// RGCN_51488067944932
// MI455X (gfx1250) — matrix operand order repaired and checked by reading, not compiled, not run on hardware
//
#include <hip/hip_runtime.h>
#include <hip/hip_bf16.h>
#include <math.h>

typedef _Float16 v16h __attribute__((ext_vector_type(16)));
typedef _Float16 v8h  __attribute__((ext_vector_type(8)));
typedef float    v8f  __attribute__((ext_vector_type(8)));

#define RREL 6

__device__ __forceinline__ v16h cat8(v8h lo, v8h hi) {
  return __builtin_shufflevector(lo, hi, 0,1,2,3,4,5,6,7,8,9,10,11,12,13,14,15);
}

// order-preserving float<->uint key for atomic min/max
__device__ __forceinline__ unsigned fkey(float f) {
  unsigned u = __float_as_uint(f);
  return (u & 0x80000000u) ? ~u : (u | 0x80000000u);
}
__device__ __forceinline__ float fdecode(unsigned k) {
  return __uint_as_float((k & 0x80000000u) ? (k & 0x7FFFFFFFu) : ~k);
}

// ---------------- weight / data packing ----------------

// Pack [R,K,H] relation weights + [K,H] root into WMMA B-operand layout.
// Logical B is [K, Mout] with Mout = (R+1)*H; columns [r*H .. r*H+H) are
// relation r, columns [R*H ..) are the root transform.
// Packed layout: [colTile][kb][lane][16 halves], lane: N = lane&15,
// element e: k = kb*32 + 16*(lane>>4) + e   (contiguous k order, same as the A loads)
__global__ void pack_weights(const float* __restrict__ w, const float* __restrict__ root,
                             _Float16* __restrict__ out, int K, int H) {
  int Mout = (RREL + 1) * H;
  int numKb = K >> 5;
  int total = Mout * K;
  int p = blockIdx.x * blockDim.x + threadIdx.x;
  if (p >= total) return;
  int e    = p & 15;
  int lane = (p >> 4) & 31;
  int kb   = (p >> 9) % numKb;
  int mt   = p / (512 * numKb);
  int m = mt * 16 + (lane & 15);
  int k = kb * 32 + ((lane >> 4) << 4) + e;
  float val;
  if (m < RREL * H) {
    int r = m / H, o = m % H;
    val = w[((size_t)r * K + k) * H + o];
  } else {
    val = root[(size_t)k * H + (m - RREL * H)];
  }
  out[p] = (_Float16)val;
}

// af2 [N2,16] f32 -> packed B layout per 16-node tile, K padded 16->32 with 0
__global__ void pack_af2(const float* __restrict__ af2, _Float16* __restrict__ out,
                         int n2Tiles) {
  int total = n2Tiles * 512;
  int p = blockIdx.x * blockDim.x + threadIdx.x;
  if (p >= total) return;
  int e    = p & 15;
  int lane = (p >> 4) & 31;
  int jT   = p >> 9;
  int j = jT * 16 + (lane & 15);
  int k = ((lane >> 4) << 4) + e;
  float v = (k < 16) ? af2[(size_t)j * 16 + k] : 0.0f;
  out[p] = (_Float16)v;
}

// af1 [N1,16] f32 -> row-major f16 [N1,32], K padded with zeros
__global__ void pad_af1h(const float* __restrict__ af1, _Float16* __restrict__ out, int N) {
  int p = blockIdx.x * blockDim.x + threadIdx.x;
  if (p >= N * 32) return;
  int n = p >> 5, k = p & 31;
  out[p] = (_Float16)((k < 16) ? af1[(size_t)n * 16 + k] : 0.0f);
}

__global__ void f32_to_f16(const float* __restrict__ in, _Float16* __restrict__ out, int n) {
  int p = blockIdx.x * blockDim.x + threadIdx.x;
  if (p < n) out[p] = (_Float16)in[p];
}

__global__ void init_misc(unsigned* mm, unsigned* bins, float* e1s, float* e2s) {
  int t = threadIdx.x;
  if (t == 0) { mm[0] = 0xFFFFFFFFu; mm[1] = 0u; }
  if (t < 16) { bins[t] = 0u; e1s[t] = 0.0f; e2s[t] = 0.0f; }
}

// ---------------- WMMA GEMM: C[N,Mout] = A(f16)[N,K] x Bpacked(f16)[K,Mout] ----------------
__global__ void gemm_wmma(const _Float16* __restrict__ A, const _Float16* __restrict__ B,
                          float* __restrict__ C, int N, int K, int Mout) {
  int lane = threadIdx.x & 31;
  int wave = threadIdx.x >> 5;
  int rowTiles = N >> 4;
  int tilesM   = Mout >> 4;
  int total = rowTiles * tilesM;
  int t = blockIdx.x * (blockDim.x >> 5) + wave;   // wave-uniform
  if (t >= total) return;                          // wave-uniform exit, EXEC stays full
  int rowTile = t % rowTiles;
  int colTile = t / rowTiles;
  int numKb = K >> 5;

  // A fragment source: row M = rowTile*16 + (lane&15); K-half select by lane>=16
  const _Float16* arow  = A + (size_t)(rowTile * 16 + (lane & 15)) * K + ((lane >> 4) << 4);
  const _Float16* bbase = B + ((size_t)colTile * numKb * 32 + lane) * 16;

  v8f acc = {};
#pragma unroll 4
  for (int kb = 0; kb < numKb; ++kb) {
    v8h alo = *(const v8h*)(arow + kb * 32);        // K = kb*32 + 16*(lane>=16) + 0..7
    v8h ahi = *(const v8h*)(arow + kb * 32 + 8);    // K = +8
    v8h blo = *(const v8h*)(bbase + (size_t)kb * 512);
    v8h bhi = *(const v8h*)(bbase + (size_t)kb * 512 + 8);
    v16h a = cat8(alo, ahi);
    v16h b = cat8(blo, bhi);
    acc = __builtin_amdgcn_wmma_f32_16x16x32_f16(false, a, false, b,
                                                 (short)0, acc, false, false);
  }
  // C/D layout: lane -> col N = lane&15; VGPR v -> row M = v + 8*(lane>=16)
  int rbase = rowTile * 16 + ((lane >> 4) << 3);
  int col   = colTile * 16 + (lane & 15);
#pragma unroll
  for (int v = 0; v < 8; ++v)
    C[(size_t)(rbase + v) * Mout + col] = acc[v];
}

// ---------------- edge aggregation ----------------
// One thread per (edge, 4-channel chunk): float4 load (coalesced across the
// chunk dimension) + 4 adjacent global f32 atomics. Maximizes outstanding
// atomic traffic to L2 instead of 64 serial atomics per thread.
__global__ void edge_agg(const int* __restrict__ ei, const int* __restrict__ et,
                         const float* __restrict__ xwcat,
                         float* __restrict__ sums, unsigned* __restrict__ cnt,
                         int E, int H, int Mout) {
  int chunks = H >> 2;
  long long idx = (long long)blockIdx.x * blockDim.x + threadIdx.x;
  long long total = (long long)E * chunks;
  if (idx >= total) return;
  int e  = (int)(idx / chunks);
  int c4 = (int)(idx % chunks) << 2;
  int src = ei[e];
  int dst = ei[E + e];
  int r   = et[e];
  const float4 m = *(const float4*)(xwcat + (size_t)src * Mout + r * H + c4);
  float* s = sums + ((size_t)dst * RREL + r) * H + c4;
  atomicAdd(&s[0], m.x);
  atomicAdd(&s[1], m.y);
  atomicAdd(&s[2], m.z);
  atomicAdd(&s[3], m.w);
  if (c4 == 0) atomicAdd(&cnt[dst * RREL + r], 1u);
}

__global__ void combine(const float* __restrict__ sums, const unsigned* __restrict__ cnt,
                        const float* __restrict__ xwcat, const float* __restrict__ bias,
                        int N, int H, int Mout, int relu,
                        _Float16* __restrict__ xh_out, float* __restrict__ af_out) {
  int idx = blockIdx.x * blockDim.x + threadIdx.x;
  if (idx >= N * H) return;
  int n = idx / H, c = idx % H;
  float s = 0.0f;
#pragma unroll
  for (int r = 0; r < RREL; ++r) {
    float cc = (float)cnt[n * RREL + r];
    s += sums[((size_t)n * RREL + r) * H + c] / (cc > 0.0f ? cc : 1.0f);
  }
  s += xwcat[(size_t)n * Mout + RREL * H + c] + bias[c];
  if (relu) s = fmaxf(s, 0.0f);
  if (xh_out) xh_out[idx] = (_Float16)s;
  if (af_out) af_out[idx] = s;
}

__global__ void mean_accum(const float* __restrict__ af, int N, float* __restrict__ esum) {
  __shared__ float sb[16];
  if (threadIdx.x < 16) sb[threadIdx.x] = 0.0f;
  __syncthreads();
  for (int n = blockIdx.x * blockDim.x + threadIdx.x; n < N; n += gridDim.x * blockDim.x) {
#pragma unroll
    for (int k = 0; k < 16; ++k) atomicAdd(&sb[k], af[(size_t)n * 16 + k]);
  }
  __syncthreads();
  if (threadIdx.x < 16) atomicAdd(&esum[threadIdx.x], sb[threadIdx.x]);
}

// ---------------- score passes: one WMMA per 16x16 tile of af1 @ af2^T ----------------
__global__ void score_minmax(const _Float16* __restrict__ af1h, const _Float16* __restrict__ af2pk,
                             int n1Tiles, int n2Tiles, unsigned* __restrict__ mm) {
  __shared__ float smin[256], smax[256];
  int lane = threadIdx.x & 31;
  int wave = threadIdx.x >> 5;
  long long t = (long long)blockIdx.x * (blockDim.x >> 5) + wave;  // wave-uniform
  long long total = (long long)n1Tiles * n2Tiles;
  float lmin = __builtin_inff(), lmax = -__builtin_inff();
  if (t < total) {
    int iT = (int)(t % n1Tiles);
    int jT = (int)(t / n1Tiles);
    const _Float16* arow = af1h + (size_t)(iT * 16 + (lane & 15)) * 32 + ((lane >> 4) << 4);
    v8h alo = *(const v8h*)arow;
    v8h ahi = *(const v8h*)(arow + 8);
    const _Float16* bp = af2pk + ((size_t)jT * 32 + lane) * 16;
    v8h blo = *(const v8h*)bp;
    v8h bhi = *(const v8h*)(bp + 8);
    v8f acc = {};
    acc = __builtin_amdgcn_wmma_f32_16x16x32_f16(false, cat8(alo, ahi), false, cat8(blo, bhi),
                                                 (short)0, acc, false, false);
#pragma unroll
    for (int v = 0; v < 8; ++v) { lmin = fminf(lmin, acc[v]); lmax = fmaxf(lmax, acc[v]); }
  }
  smin[threadIdx.x] = lmin; smax[threadIdx.x] = lmax;
  __syncthreads();
  for (int s = 128; s > 0; s >>= 1) {
    if (threadIdx.x < s) {
      smin[threadIdx.x] = fminf(smin[threadIdx.x], smin[threadIdx.x + s]);
      smax[threadIdx.x] = fmaxf(smax[threadIdx.x], smax[threadIdx.x + s]);
    }
    __syncthreads();
  }
  if (threadIdx.x == 0) {
    atomicMin(&mm[0], fkey(smin[0]));
    atomicMax(&mm[1], fkey(smax[0]));
  }
}

__global__ void score_hist(const _Float16* __restrict__ af1h, const _Float16* __restrict__ af2pk,
                           int n1Tiles, int n2Tiles, const unsigned* __restrict__ mm,
                           unsigned* __restrict__ bins) {
  __shared__ unsigned shb[16];
  if (threadIdx.x < 16) shb[threadIdx.x] = 0u;
  __syncthreads();
  int lane = threadIdx.x & 31;
  int wave = threadIdx.x >> 5;
  long long t = (long long)blockIdx.x * (blockDim.x >> 5) + wave;  // wave-uniform
  long long total = (long long)n1Tiles * n2Tiles;
  if (t < total) {
    float mn = fdecode(mm[0]);
    float mx = fdecode(mm[1]);
    float width = (mx > mn) ? (mx - mn) / 16.0f : 1.0f;
    int iT = (int)(t % n1Tiles);
    int jT = (int)(t / n1Tiles);
    const _Float16* arow = af1h + (size_t)(iT * 16 + (lane & 15)) * 32 + ((lane >> 4) << 4);
    v8h alo = *(const v8h*)arow;
    v8h ahi = *(const v8h*)(arow + 8);
    const _Float16* bp = af2pk + ((size_t)jT * 32 + lane) * 16;
    v8h blo = *(const v8h*)bp;
    v8h bhi = *(const v8h*)(bp + 8);
    v8f acc = {};
    acc = __builtin_amdgcn_wmma_f32_16x16x32_f16(false, cat8(alo, ahi), false, cat8(blo, bhi),
                                                 (short)0, acc, false, false);
#pragma unroll
    for (int v = 0; v < 8; ++v) {
      int b = (int)floorf((acc[v] - mn) / width);
      b = b < 0 ? 0 : (b > 15 ? 15 : b);
      atomicAdd(&shb[b], 1u);
    }
  }
  __syncthreads();
  if (threadIdx.x < 16) atomicAdd(&bins[threadIdx.x], shb[threadIdx.x]);
}

// ---------------- tiny head ----------------
__global__ void final_head(const float* e1s, const float* e2s, int N1, int N2,
                           const unsigned* bins,
                           const float* ntn_w, const float* ntn_wb, const float* ntn_b,
                           const float* fc1_w, const float* fc1_b,
                           const float* score_w, const float* score_b, float* out) {
  if (threadIdx.x != 0 || blockIdx.x != 0) return;
  float e1[16], e2[16];
  for (int i = 0; i < 16; ++i) { e1[i] = e1s[i] / (float)N1; e2[i] = e2s[i] / (float)N2; }
  float sim[32];
  for (int tt = 0; tt < 16; ++tt) {
    float sc = 0.0f;
    for (int i = 0; i < 16; ++i)
      for (int j = 0; j < 16; ++j)
        sc += e1[i] * ntn_w[((size_t)i * 16 + j) * 16 + tt] * e2[j];
    float blk = 0.0f;
    for (int k = 0; k < 32; ++k)
      blk += ntn_wb[(size_t)tt * 32 + k] * (k < 16 ? e1[k] : e2[k - 16]);
    float v = sc + blk + ntn_b[tt];
    sim[tt] = v > 0.0f ? v : 0.0f;
  }
  double tot = (double)N1 * (double)N2;
  for (int i = 0; i < 16; ++i) sim[16 + i] = (float)((double)bins[i] / tot);
  float h[16];
  for (int j = 0; j < 16; ++j) {
    float a = fc1_b[j];
    for (int k = 0; k < 32; ++k) a += sim[k] * fc1_w[(size_t)k * 16 + j];
    h[j] = tanhf(a);
  }
  float logit = score_b[0];
  for (int j = 0; j < 16; ++j) logit += h[j] * score_w[j];
  out[0] = 1.0f / (1.0f + expf(-logit));
  out[1] = logit;
}

// ---------------- launch ----------------
extern "C" void kernel_launch(void* const* d_in, const int* in_sizes, int n_in,
                              void* d_out, int out_size, void* d_ws, size_t ws_size,
                              hipStream_t stream) {
  (void)n_in; (void)out_size; (void)ws_size;
  const float* feat1 = (const float*)d_in[0];
  const float* feat2 = (const float*)d_in[1];
  const int*   ei1   = (const int*)d_in[2];
  const int*   et1   = (const int*)d_in[3];
  const int*   ei2   = (const int*)d_in[4];
  const int*   et2   = (const int*)d_in[5];
  const float* w1 = (const float*)d_in[6];
  const float* r1 = (const float*)d_in[7];
  const float* b1 = (const float*)d_in[8];
  const float* w2 = (const float*)d_in[9];
  const float* r2 = (const float*)d_in[10];
  const float* b2 = (const float*)d_in[11];
  const float* w3 = (const float*)d_in[12];
  const float* r3 = (const float*)d_in[13];
  const float* b3 = (const float*)d_in[14];
  const float* ntn_w  = (const float*)d_in[15];
  const float* ntn_wb = (const float*)d_in[16];
  const float* ntn_b  = (const float*)d_in[17];
  const float* fc1_w  = (const float*)d_in[18];
  const float* fc1_b  = (const float*)d_in[19];
  const float* sc_w   = (const float*)d_in[20];
  const float* sc_b   = (const float*)d_in[21];
  float* out = (float*)d_out;

  const int IN = 128, HH1 = 64, HH2 = 32, OO = 16;
  const int M1 = (RREL + 1) * HH1;  // 448
  const int M2 = (RREL + 1) * HH2;  // 224
  const int M3 = (RREL + 1) * OO;   // 112
  int N1 = in_sizes[0] / IN;        // 10000 (multiple of 16)
  int N2 = in_sizes[1] / IN;
  int E1 = in_sizes[3];
  int E2 = in_sizes[5];
  int Nmax = N1 > N2 ? N1 : N2;

  size_t off = 0;
  auto alloc = [&](size_t b) { size_t o = off; off = (off + b + 255) & ~(size_t)255; return o; };
  char* ws = (char*)d_ws;
  _Float16* WP1   = (_Float16*)(ws + alloc((size_t)IN * M1 * 2));
  _Float16* WP2   = (_Float16*)(ws + alloc((size_t)HH1 * M2 * 2));
  _Float16* WP3   = (_Float16*)(ws + alloc((size_t)HH2 * M3 * 2));
  _Float16* XH    = (_Float16*)(ws + alloc((size_t)Nmax * IN * 2));
  float*    XWCAT = (float*)   (ws + alloc((size_t)Nmax * M1 * 4));
  float*    SUMS  = (float*)   (ws + alloc((size_t)Nmax * RREL * HH1 * 4));
  unsigned* CNT   = (unsigned*)(ws + alloc((size_t)Nmax * RREL * 4));
  float*    AF1   = (float*)   (ws + alloc((size_t)N1 * OO * 4));
  float*    AF2   = (float*)   (ws + alloc((size_t)N2 * OO * 4));
  _Float16* AF1H  = (_Float16*)(ws + alloc((size_t)N1 * 32 * 2));
  _Float16* AF2PK = (_Float16*)(ws + alloc((size_t)(N2 / 16) * 512 * 2));
  float*    E1S   = (float*)   (ws + alloc(16 * 4));
  float*    E2S   = (float*)   (ws + alloc(16 * 4));
  unsigned* MM    = (unsigned*)(ws + alloc(2 * 4));
  unsigned* HB    = (unsigned*)(ws + alloc(16 * 4));

  auto cdiv = [](long long a, long long b) { return (int)((a + b - 1) / b); };

  // pack weights (once per call, deterministic)
  pack_weights<<<cdiv((long long)IN * M1, 256), 256, 0, stream>>>(w1, r1, WP1, IN, HH1);
  pack_weights<<<cdiv((long long)HH1 * M2, 256), 256, 0, stream>>>(w2, r2, WP2, HH1, HH2);
  pack_weights<<<cdiv((long long)HH2 * M3, 256), 256, 0, stream>>>(w3, r3, WP3, HH2, OO);
  init_misc<<<1, 32, 0, stream>>>(MM, HB, E1S, E2S);

  const float* feats[2] = {feat1, feat2};
  const int*   eis[2]   = {ei1, ei2};
  const int*   ets[2]   = {et1, et2};
  int Ns[2] = {N1, N2};
  int Es[2] = {E1, E2};
  float* afs[2]   = {AF1, AF2};
  float* esums[2] = {E1S, E2S};

  struct LayerCfg { const _Float16* wp; const float* bias; int K, H, Mout, relu; };
  LayerCfg layers[3] = {
    {WP1, b1, IN,  HH1, M1, 1},
    {WP2, b2, HH1, HH2, M2, 1},
    {WP3, b3, HH2, OO,  M3, 0},
  };

  for (int g = 0; g < 2; ++g) {
    int N = Ns[g], E = Es[g];
    f32_to_f16<<<cdiv((long long)N * IN, 256), 256, 0, stream>>>(feats[g], XH, N * IN);
    for (int L = 0; L < 3; ++L) {
      const LayerCfg& lc = layers[L];
      long long waves = (long long)(N / 16) * (lc.Mout / 16);
      gemm_wmma<<<cdiv(waves, 8), 256, 0, stream>>>(XH, lc.wp, XWCAT, N, lc.K, lc.Mout);
      hipMemsetAsync(SUMS, 0, (size_t)N * RREL * lc.H * 4, stream);
      hipMemsetAsync(CNT, 0, (size_t)N * RREL * 4, stream);
      long long eitems = (long long)E * (lc.H >> 2);
      edge_agg<<<cdiv(eitems, 256), 256, 0, stream>>>(eis[g], ets[g], XWCAT, SUMS, CNT, E, lc.H, lc.Mout);
      combine<<<cdiv((long long)N * lc.H, 256), 256, 0, stream>>>(
          SUMS, CNT, XWCAT, lc.bias, N, lc.H, lc.Mout, lc.relu,
          (L < 2) ? XH : (_Float16*)nullptr, (L < 2) ? (float*)nullptr : afs[g]);
    }
    mean_accum<<<64, 256, 0, stream>>>(afs[g], N, esums[g]);
  }

  pad_af1h<<<cdiv((long long)N1 * 32, 256), 256, 0, stream>>>(AF1, AF1H, N1);
  pack_af2<<<cdiv((long long)(N2 / 16) * 512, 256), 256, 0, stream>>>(AF2, AF2PK, N2 / 16);

  long long tiles = (long long)(N1 / 16) * (N2 / 16);
  score_minmax<<<cdiv(tiles, 8), 256, 0, stream>>>(AF1H, AF2PK, N1 / 16, N2 / 16, MM);
  score_hist<<<cdiv(tiles, 8), 256, 0, stream>>>(AF1H, AF2PK, N1 / 16, N2 / 16, MM, HB);

  final_head<<<1, 1, 0, stream>>>(E1S, E2S, N1, N2, HB,
                                  ntn_w, ntn_wb, ntn_b, fc1_w, fc1_b, sc_w, sc_b, out);
}
